// Embeddings_23381801959964
// MI455X (gfx1250) — matrix operand order repaired and checked by reading, not compiled, not run on hardware
//
#include <hip/hip_runtime.h>
#include <hip/hip_bf16.h>
#include <cstdint>
#include <cstddef>

// Problem constants (from reference)
#define B_   64
#define T_   512
#define HB   768
#define WMAX 256
#define CMAX 16
#define E_   64
#define R_   256
#define G4   1024   // 4*R gates

#define MTILES 2
#define MSEQ   (MTILES * 16)   // 32 sequences per block

typedef __bf16 bf16_t;
typedef bf16_t v16bf __attribute__((ext_vector_type(16)));
typedef float  v8f   __attribute__((ext_vector_type(8)));

union FragU { uint4 q[2]; v16bf v; };

__device__ __forceinline__ unsigned short f2bf(float x) {
    unsigned int u = __float_as_uint(x);
    u += 0x7FFFu + ((u >> 16) & 1u);   // round-to-nearest-even
    return (unsigned short)(u >> 16);
}

// 16-half fragment as two 16B pieces separated by `gap` halves.
// A fragment: gap=8 too (contiguous, same K order as B). B fragment: gap=8 (contiguous).
__device__ __forceinline__ v16bf frag_pair(const unsigned short* p, int gap) {
    FragU f;
    f.q[0] = *(const uint4*)(p);
    f.q[1] = *(const uint4*)(p + gap);
    return f.v;
}

// ---------------------------------------------------------------- utilities
__global__ void zero_kernel(float* p, size_t n) {
    size_t i = (size_t)blockIdx.x * blockDim.x + threadIdx.x;
    size_t stride = (size_t)gridDim.x * blockDim.x;
    for (; i < n; i += stride) p[i] = 0.0f;
}

// f32 -> bf16 weight copies + fused bias
__global__ void convert_kernel(const float* __restrict__ wih, const float* __restrict__ whh,
                               const float* __restrict__ linw, const float* __restrict__ bih,
                               const float* __restrict__ bhh,
                               unsigned short* __restrict__ wih_b, unsigned short* __restrict__ whh_b,
                               unsigned short* __restrict__ linw_b, float* __restrict__ bias) {
    const int N0 = G4 * E_;            // 65536
    const int N1 = G4 * R_;            // 262144
    const int N2 = HB * R_;            // 196608
    const int NT = N0 + N1 + N2 + G4;
    int i = blockIdx.x * blockDim.x + threadIdx.x;
    int stride = gridDim.x * blockDim.x;
    for (; i < NT; i += stride) {
        if (i < N0)                wih_b[i] = f2bf(wih[i]);
        else if (i < N0 + N1)      whh_b[i - N0] = f2bf(whh[i - N0]);
        else if (i < N0 + N1 + N2) linw_b[i - N0 - N1] = f2bf(linw[i - N0 - N1]);
        else { int j = i - N0 - N1 - N2; bias[j] = bih[j] + bhh[j]; }
    }
}

// Gather char embeddings into xbuf[seq][c][e] (bf16). One thread = 16 halves.
__global__ void gather_chars(const int* __restrict__ char_ids, const float* __restrict__ cemb,
                             unsigned short* __restrict__ xbuf) {
    const int t  = blockIdx.x * blockDim.x + threadIdx.x;  // 0 .. 1048575
    const int e0 = (t & 3) * 16;
    const int sc = t >> 2;            // seq*16 + c
    const int c  = sc & 15;
    const int seq = sc >> 4;
    const int b = seq >> 8, w = seq & 255;
    const int id = char_ids[(size_t)b * (WMAX * CMAX) + w * CMAX + c];
    unsigned short* dst = xbuf + (size_t)sc * E_ + e0;
    if (id < 0) {
        #pragma unroll
        for (int i = 0; i < 16; ++i) dst[i] = 0;
    } else {
        const float* er = cemb + (size_t)id * E_ + e0;
        #pragma unroll
        for (int i = 0; i < 16; ++i) dst[i] = f2bf(er[i]);
    }
}

// Per-batch stable compaction map: content token p -> compact index j,
// scatter bin = token_word_ids[b, j]; non-content -> -1.
__global__ __launch_bounds__(512)
void content_scan(const int* __restrict__ am, const int* __restrict__ sp,
                  const int* __restrict__ twi, int* __restrict__ widmap) {
    __shared__ int s[T_];
    const int b = blockIdx.x, t = threadIdx.x;
    const int c = (am[b * T_ + t] != 0 && sp[b * T_ + t] == 0) ? 1 : 0;
    s[t] = c;
    __syncthreads();
    for (int off = 1; off < T_; off <<= 1) {
        int v = (t >= off) ? s[t - off] : 0;
        __syncthreads();
        s[t] += v;
        __syncthreads();
    }
    const int j = s[t] - c;  // exclusive prefix
    widmap[b * T_ + t] = c ? twi[b * T_ + j] : -1;
}

// Scatter-add compacted hidden states into word bins (hw f32 atomics).
__global__ void bert_scatter(const float* __restrict__ lh, const int* __restrict__ widmap,
                             float* __restrict__ out) {
    const int p = blockIdx.x;
    const int b = blockIdx.y;
    const int wid = widmap[b * T_ + p];
    if (wid < 0) return;
    const float* src = lh + ((size_t)b * T_ + p) * HB;
    float* dst = out + ((size_t)b * WMAX + wid) * HB;
    for (int h = threadIdx.x; h < HB; h += blockDim.x)
        __hip_atomic_fetch_add(&dst[h], src[h], __ATOMIC_RELAXED, __HIP_MEMORY_SCOPE_AGENT);
}

// ---------------------------------------------------------------- WMMA LSTM
// Block = 32 sequences, 512 threads = 16 waves. Wave w owns hidden tile r-range
// [w*16, w*16+16) i.e. gate N-tiles {w, w+16, w+32, w+48}: each lane then holds
// i,f,g,o for its own hidden column r, so the cell update is done in registers
// in the WMMA D-fragment layout. Only h round-trips LDS (bf16, double buffered).
__global__ __launch_bounds__(512)
void lstm_kernel(const unsigned short* __restrict__ xbuf,  // [seq][c][e] bf16
                 const unsigned short* __restrict__ wih,   // (1024,64) bf16 K-contig
                 const unsigned short* __restrict__ whh,   // (1024,256) bf16
                 const float* __restrict__ bias,           // (1024) b_ih+b_hh
                 const unsigned short* __restrict__ linw,  // (768,256) bf16
                 const float* __restrict__ linb,           // (768)
                 float* __restrict__ out) {                // (B*W, 768) accumulate
    extern __shared__ char smem[];
    unsigned short* hb0 = (unsigned short*)smem;            // 32*256 bf16 = 16KB
    unsigned short* hb1 = (unsigned short*)(smem + 16384);  // 16KB

    const int tid  = threadIdx.x;
    const int lane = tid & 31;
    const int wv   = tid >> 5;          // 0..15 == r-tile index
    const int seq0 = blockIdx.x * MSEQ;

    const int nlane = lane & 15;
    const int hi    = lane >> 4;
    const int aoff  = hi ? 16 : 0;      // A fragment K offset in K-chunk
    const int boff  = hi ? 16 : 0;      // B fragment K offset in K-chunk
    const int r     = wv * 16 + nlane;  // this lane's hidden column

    // zero h0 (16KB = 2048 u64)
    {
        unsigned long long* p = (unsigned long long*)hb0;
        #pragma unroll
        for (int i = 0; i < 4; ++i) p[tid + i * 512] = 0ull;
    }

    float cst[MTILES][8];
    #pragma unroll
    for (int mt = 0; mt < MTILES; ++mt)
        #pragma unroll
        for (int v = 0; v < 8; ++v) cst[mt][v] = 0.f;

    float bg[4];
    #pragma unroll
    for (int g = 0; g < 4; ++g) bg[g] = bias[g * R_ + r];

    __syncthreads();

    #pragma clang loop unroll(disable)
    for (int c = 0; c < CMAX; ++c) {
        // Defeat LICM: weight fragments must be re-streamed from L2 each step,
        // not hoisted into (and spilled out of) registers.
        asm volatile("" ::: "memory");
        const unsigned short* hcur = (c & 1) ? hb1 : hb0;
        unsigned short*       hnxt = (c & 1) ? hb0 : hb1;

        v8f acc[4][MTILES];
        #pragma unroll
        for (int g = 0; g < 4; ++g)
            #pragma unroll
            for (int mt = 0; mt < MTILES; ++mt)
                #pragma unroll
                for (int i = 0; i < 8; ++i) acc[g][mt][i] = bg[g];

        // x_t contribution: K = 64 -> 2 chunks, A straight from global xbuf
        #pragma unroll
        for (int kc = 0; kc < 2; ++kc) {
            v16bf a[MTILES];
            #pragma unroll
            for (int mt = 0; mt < MTILES; ++mt) {
                const int seq = seq0 + mt * 16 + nlane;
                a[mt] = frag_pair(xbuf + (size_t)seq * (CMAX * E_) + c * E_ + kc * 32 + aoff, 8);
            }
            #pragma unroll
            for (int g = 0; g < 4; ++g) {
                const int n = g * R_ + r;
                v16bf b = frag_pair(wih + (size_t)n * E_ + kc * 32 + boff, 8);
                #pragma unroll
                for (int mt = 0; mt < MTILES; ++mt)
                    acc[g][mt] = __builtin_amdgcn_wmma_f32_16x16x32_bf16(
                        false, a[mt], false, b, (short)0, acc[g][mt], false, false);
            }
        }
        // h contribution: K = 256 -> 8 chunks, A from LDS
        #pragma unroll
        for (int kc = 0; kc < 8; ++kc) {
            v16bf a[MTILES];
            #pragma unroll
            for (int mt = 0; mt < MTILES; ++mt)
                a[mt] = frag_pair(hcur + (mt * 16 + nlane) * R_ + kc * 32 + aoff, 8);
            #pragma unroll
            for (int g = 0; g < 4; ++g) {
                const int n = g * R_ + r;
                v16bf b = frag_pair(whh + (size_t)n * R_ + kc * 32 + boff, 8);
                #pragma unroll
                for (int mt = 0; mt < MTILES; ++mt)
                    acc[g][mt] = __builtin_amdgcn_wmma_f32_16x16x32_bf16(
                        false, a[mt], false, b, (short)0, acc[g][mt], false, false);
            }
        }
        // fused LSTM cell update, entirely in D-fragment layout
        #pragma unroll
        for (int mt = 0; mt < MTILES; ++mt) {
            #pragma unroll
            for (int v = 0; v < 8; ++v) {
                const float gi = acc[0][mt][v];
                const float gf = acc[1][mt][v];
                const float gg = acc[2][mt][v];
                const float go = acc[3][mt][v];
                const float si = 1.f / (1.f + expf(-gi));
                const float sf = 1.f / (1.f + expf(-gf));
                const float so = 1.f / (1.f + expf(-go));
                const float cc = sf * cst[mt][v] + si * tanhf(gg);
                cst[mt][v] = cc;
                hnxt[(mt * 16 + v + 8 * hi) * R_ + r] = f2bf(so * tanhf(cc));
            }
        }
        __syncthreads();
    }

    // 16 steps: last write (c=15) went to hb0
    const unsigned short* hf = hb0;

    // Final linear: (32 x 256) @ (256 x 768) + lin_b, accumulated into out.
    // 48 N-tiles; wave w handles tiles {3w, 3w+1, 3w+2}.
    #pragma unroll
    for (int nt3 = 0; nt3 < 3; ++nt3) {
        asm volatile("" ::: "memory");
        const int nt = wv * 3 + nt3;
        const int n  = nt * 16 + nlane;
        const float bv = linb[n];
        v8f acc[MTILES];
        #pragma unroll
        for (int mt = 0; mt < MTILES; ++mt)
            #pragma unroll
            for (int i = 0; i < 8; ++i) acc[mt][i] = bv;
        #pragma unroll
        for (int kc = 0; kc < 8; ++kc) {
            v16bf b = frag_pair(linw + (size_t)n * R_ + kc * 32 + boff, 8);
            #pragma unroll
            for (int mt = 0; mt < MTILES; ++mt) {
                v16bf a = frag_pair(hf + (mt * 16 + nlane) * R_ + kc * 32 + aoff, 8);
                acc[mt] = __builtin_amdgcn_wmma_f32_16x16x32_bf16(
                    false, a, false, b, (short)0, acc[mt], false, false);
            }
        }
        #pragma unroll
        for (int mt = 0; mt < MTILES; ++mt)
            #pragma unroll
            for (int v = 0; v < 8; ++v) {
                const size_t idx = (size_t)(seq0 + mt * 16 + v + 8 * hi) * HB + n;
                out[idx] += acc[mt][v];
            }
    }
}

// word_level_mask = any(word_embeds != 0) along H, stored as 0/1 float
__global__ void mask_kernel(const float* __restrict__ we, float* __restrict__ mask) {
    const int pair = blockIdx.x * blockDim.x + threadIdx.x;
    if (pair >= B_ * WMAX) return;
    const float4* p = (const float4*)(we + (size_t)pair * HB);
    int any = 0;
    for (int i = 0; i < HB / 4; ++i) {
        float4 v = p[i];
        any |= (v.x != 0.f) | (v.y != 0.f) | (v.z != 0.f) | (v.w != 0.f);
    }
    mask[pair] = any ? 1.0f : 0.0f;
}

// ---------------------------------------------------------------- launch
extern "C" void kernel_launch(void* const* d_in, const int* in_sizes, int n_in,
                              void* d_out, int out_size, void* d_ws, size_t ws_size,
                              hipStream_t stream) {
    const int*   char_ids = (const int*)d_in[0];
    // d_in[1] = char_word_ids (unused by reference)
    const int*   twi      = (const int*)d_in[2];
    const int*   am       = (const int*)d_in[3];
    const int*   sp       = (const int*)d_in[4];
    const float* lh       = (const float*)d_in[5];
    const float* cemb     = (const float*)d_in[6];
    const float* wih      = (const float*)d_in[7];
    const float* whh      = (const float*)d_in[8];
    const float* bih      = (const float*)d_in[9];
    const float* bhh      = (const float*)d_in[10];
    const float* linw     = (const float*)d_in[11];
    const float* linb     = (const float*)d_in[12];
    float* out = (float*)d_out;

    char* ws = (char*)d_ws;
    int*            widmap = (int*)ws;                            //  131072 B
    unsigned short* wih_b  = (unsigned short*)(ws + 131072);      //  131072 B
    unsigned short* whh_b  = (unsigned short*)(ws + 262144);      //  524288 B
    unsigned short* linw_b = (unsigned short*)(ws + 786432);      //  393216 B
    float*          bias   = (float*)(ws + 1179648);              //    4096 B
    unsigned short* xbuf   = (unsigned short*)(ws + 1183744);     // 33554432 B

    const size_t n_we = (size_t)B_ * WMAX * HB;

    zero_kernel<<<2048, 256, 0, stream>>>(out, n_we + (size_t)B_ * WMAX);
    convert_kernel<<<512, 256, 0, stream>>>(wih, whh, linw, bih, bhh,
                                            wih_b, whh_b, linw_b, bias);
    gather_chars<<<4096, 256, 0, stream>>>(char_ids, cemb, xbuf);
    content_scan<<<B_, T_, 0, stream>>>(am, sp, twi, widmap);
    dim3 gscat(T_, B_);
    bert_scatter<<<gscat, 256, 0, stream>>>(lh, widmap, out);
    lstm_kernel<<<(B_ * WMAX) / MSEQ, 512, 32768, stream>>>(
        xbuf, wih_b, whh_b, bias, linw_b, linb, out);
    mask_kernel<<<(B_ * WMAX) / 256, 256, 0, stream>>>(out, out + n_we);
}
